// MultiHeadAttention_25013889532079
// MI455X (gfx1250) — matrix operand order repaired and checked by reading, not compiled, not run on hardware
//
#include <hip/hip_runtime.h>

// ---------------------------------------------------------------------------
// MHA forward for MI455X (gfx1250, wave32, WMMA 16x16x32 f16 -> f32)
// B=2, S=2048, D_MODEL=1024, H=16, Dk=64
// Software-pipelined fragment loads to keep the matrix pipe fed.
// ---------------------------------------------------------------------------

#define DMODEL 1024
#define HEADS  16
#define DKH    64
#define BATCH  2
#define SEQ    2048
#define MROWS  (BATCH * SEQ)   // 4096 total rows for the dense GEMMs

typedef __attribute__((ext_vector_type(4)))  _Float16 v4h;
typedef __attribute__((ext_vector_type(8)))  _Float16 v8h;
typedef __attribute__((ext_vector_type(16))) _Float16 v16h;
typedef __attribute__((ext_vector_type(8)))  float    v8f;

static __device__ __forceinline__ v16h cat8(v8h lo, v8h hi) {
  return __builtin_shufflevector(lo, hi, 0,1,2,3,4,5,6,7,8,9,10,11,12,13,14,15);
}

static __device__ __forceinline__ v8f wmma16(v16h a, v16h b, v8f c) {
  // D = A(16x32 f16) * B(32x16 f16) + C(16x16 f32)
  return __builtin_amdgcn_wmma_f32_16x16x32_f16(
      /*neg_a=*/false, a, /*neg_b=*/false, b,
      /*c_mod=*/(short)0, c, /*reuse_a=*/false, /*reuse_b=*/false);
}

// A fragment (16x32, 16-bit): lane L -> row = L&15; data = two contiguous
// 8-half chunks at k0 + (L>>4)*8 and k0+16+(L>>4)*8.   rowk = &row[k0]
static __device__ __forceinline__ v16h ld_a_frag(const _Float16* rowk, int hi16) {
  const v8h lo = *(const v8h*)(rowk + hi16 * 8);
  const v8h hi = *(const v8h*)(rowk + 16 + hi16 * 8);
  return cat8(lo, hi);
}

// B fragment (32x16, 16-bit), fed from a row-major "W[n][k]" row (NT GEMM):
// lane L -> col n = L&15; data = two contiguous 8-half chunks at
// k0 + (L>>4)*8 and k0+16+(L>>4)*8 (same k order as the A fragment).
static __device__ __forceinline__ v16h ld_b_frag(const _Float16* rowk, int hi16) {
  const _Float16* p = rowk + hi16 * 8;
  return cat8(*(const v8h*)p, *(const v8h*)(p + 16));
}

// Row reductions across the 16-lane halves (rows live per-VGPR, cols per-lane).
static __device__ __forceinline__ float redmax16(float v) {
  v = fmaxf(v, __shfl_xor(v, 1, 32));
  v = fmaxf(v, __shfl_xor(v, 2, 32));
  v = fmaxf(v, __shfl_xor(v, 4, 32));
  v = fmaxf(v, __shfl_xor(v, 8, 32));
  return v;
}
static __device__ __forceinline__ float redsum16(float v) {
  v += __shfl_xor(v, 1, 32);
  v += __shfl_xor(v, 2, 32);
  v += __shfl_xor(v, 4, 32);
  v += __shfl_xor(v, 8, 32);
  return v;
}

// ---------------------------------------------------------------------------
// Kernel 0: f32 -> f16 conversion (vectorized x4)
// ---------------------------------------------------------------------------
__global__ void cvt_f32_f16_kernel(const float* __restrict__ in,
                                   _Float16* __restrict__ out, int n4) {
  int i = blockIdx.x * blockDim.x + threadIdx.x;
  if (i >= n4) return;
  float4 f = ((const float4*)in)[i];
  v4h h;
  h[0] = (_Float16)f.x; h[1] = (_Float16)f.y;
  h[2] = (_Float16)f.z; h[3] = (_Float16)f.w;
  ((v4h*)out)[i] = h;
}

// ---------------------------------------------------------------------------
// Shared GEMM core: C[m,n] = sum_k X[m,k] * W[n,k] for a 32x64 wave tile,
// double-buffered over the 32-wide K steps. acc must be zeroed by caller.
// ---------------------------------------------------------------------------
static __device__ __forceinline__ void gemm32x64_nt(
    const _Float16* __restrict__ X, const _Float16* __restrict__ W,
    int m0, int n0, int l16, int hi16, v8f acc[2][4]) {
  const _Float16* xr0 = X + (size_t)(m0 + l16) * DMODEL;
  const _Float16* xr1 = X + (size_t)(m0 + 16 + l16) * DMODEL;
  const _Float16* wr0 = W + (size_t)(n0 + 0  + l16) * DMODEL;
  const _Float16* wr1 = W + (size_t)(n0 + 16 + l16) * DMODEL;
  const _Float16* wr2 = W + (size_t)(n0 + 32 + l16) * DMODEL;
  const _Float16* wr3 = W + (size_t)(n0 + 48 + l16) * DMODEL;

  // prologue: fragments for k-step 0
  v16h a0 = ld_a_frag(xr0, hi16);
  v16h a1 = ld_a_frag(xr1, hi16);
  v16h b0 = ld_b_frag(wr0, hi16);
  v16h b1 = ld_b_frag(wr1, hi16);
  v16h b2 = ld_b_frag(wr2, hi16);
  v16h b3 = ld_b_frag(wr3, hi16);

  for (int kd = 0; kd < DMODEL - 32; kd += 32) {
    // issue next k-step's 12 loads before consuming current fragments
    const int kn = kd + 32;
    v16h na0 = ld_a_frag(xr0 + kn, hi16);
    v16h na1 = ld_a_frag(xr1 + kn, hi16);
    v16h nb0 = ld_b_frag(wr0 + kn, hi16);
    v16h nb1 = ld_b_frag(wr1 + kn, hi16);
    v16h nb2 = ld_b_frag(wr2 + kn, hi16);
    v16h nb3 = ld_b_frag(wr3 + kn, hi16);

    acc[0][0] = wmma16(a0, b0, acc[0][0]);
    acc[1][0] = wmma16(a1, b0, acc[1][0]);
    acc[0][1] = wmma16(a0, b1, acc[0][1]);
    acc[1][1] = wmma16(a1, b1, acc[1][1]);
    acc[0][2] = wmma16(a0, b2, acc[0][2]);
    acc[1][2] = wmma16(a1, b2, acc[1][2]);
    acc[0][3] = wmma16(a0, b3, acc[0][3]);
    acc[1][3] = wmma16(a1, b3, acc[1][3]);

    a0 = na0; a1 = na1;
    b0 = nb0; b1 = nb1; b2 = nb2; b3 = nb3;
  }
  // epilogue: last k-step
  acc[0][0] = wmma16(a0, b0, acc[0][0]);
  acc[1][0] = wmma16(a1, b0, acc[1][0]);
  acc[0][1] = wmma16(a0, b1, acc[0][1]);
  acc[1][1] = wmma16(a1, b1, acc[1][1]);
  acc[0][2] = wmma16(a0, b2, acc[0][2]);
  acc[1][2] = wmma16(a1, b2, acc[1][2]);
  acc[0][3] = wmma16(a0, b3, acc[0][3]);
  acc[1][3] = wmma16(a1, b3, acc[1][3]);
}

// ---------------------------------------------------------------------------
// Kernel 1: projection GEMM  (x @ W^T), scatter to head layout.
// mode 0: [B,H,S,Dk]; mode 1: transposed [B,H,Dk,S] (for V).
// ---------------------------------------------------------------------------
__global__ __launch_bounds__(128)
void proj_gemm_kernel(const _Float16* __restrict__ X,
                      const _Float16* __restrict__ W,
                      _Float16* __restrict__ Out, int mode) {
  const int lane = threadIdx.x & 31;
  const int hi16 = lane >> 4;
  const int l16  = lane & 15;
  const int wid  = (blockIdx.x * blockDim.x + threadIdx.x) >> 5;  // 0..2047
  const int m0   = (wid & 127) * 32;   // 128 M-tiles of 32
  const int n0   = (wid >> 7)  * 64;   // 16  N-tiles of 64

  v8f acc[2][4];
  const v8f vzero = {0.f,0.f,0.f,0.f,0.f,0.f,0.f,0.f};
#pragma unroll
  for (int i = 0; i < 2; ++i)
#pragma unroll
    for (int j = 0; j < 4; ++j) acc[i][j] = vzero;

  gemm32x64_nt(X, W, m0, n0, l16, hi16, acc);

  // C layout: vgpr r -> row r + hi16*8, col = l16
#pragma unroll
  for (int mt = 0; mt < 2; ++mt)
#pragma unroll
    for (int nt = 0; nt < 4; ++nt)
#pragma unroll
      for (int r = 0; r < 8; ++r) {
        int m = m0 + mt * 16 + r + hi16 * 8;
        int n = n0 + nt * 16 + l16;
        int bb = m >> 11, s = m & (SEQ - 1);
        int h = n >> 6, dk = n & (DKH - 1);
        size_t idx = (mode == 0)
            ? (((size_t)(bb * HEADS + h) * SEQ + s) * DKH + dk)
            : (((size_t)(bb * HEADS + h) * DKH + dk) * SEQ + s);
        Out[idx] = (_Float16)acc[mt][nt][r];
      }
}

// ---------------------------------------------------------------------------
// Kernel 2: flash-style attention. One wave owns 16 query rows of one (b,h),
// streams keys in blocks of 64, online softmax, P transposed through LDS.
// Q,K: [B*H, S, Dk] f16.  Vt: [B*H, Dk, S] f16 (pre-transposed).
// O:  [B*S, D_MODEL] f16 (input layout for the output projection).
// ---------------------------------------------------------------------------
__global__ __launch_bounds__(128)
void attention_kernel(const _Float16* __restrict__ Q,
                      const _Float16* __restrict__ K,
                      const _Float16* __restrict__ Vt,
                      _Float16* __restrict__ O) {
  __shared__ _Float16 ldsP[4][16 * 64];   // per-wave P staging (2 KB each)

  const int lane   = threadIdx.x & 31;
  const int hi16   = lane >> 4;
  const int l16    = lane & 15;
  const int wlocal = threadIdx.x >> 5;
  const int wid    = blockIdx.x * 4 + wlocal;   // 0..4095
  const int qblk   = wid & 127;                 // 128 q-tiles of 16
  const int bh     = wid >> 7;                  // 0..31
  const int q0     = qblk * 16;

  const _Float16* Qh = Q  + (size_t)bh * SEQ * DKH;
  const _Float16* Kh = K  + (size_t)bh * SEQ * DKH;
  const _Float16* Vh = Vt + (size_t)bh * DKH * SEQ;

  // Q fragments for this wave's 16 rows (K = Dk = 64 -> two 32-wide frags)
  const _Float16* qrow = Qh + (size_t)(q0 + l16) * DKH;
  v16h qa0 = ld_a_frag(qrow + 0,  hi16);
  v16h qa1 = ld_a_frag(qrow + 32, hi16);

  const v8f vzero = {0.f,0.f,0.f,0.f,0.f,0.f,0.f,0.f};
  v8f o[4] = {vzero, vzero, vzero, vzero};      // 16 x 64 output accumulator
  float m_i[8], l_i[8];
#pragma unroll
  for (int r = 0; r < 8; ++r) { m_i[r] = -1e30f; l_i[r] = 0.f; }

  for (int jb = 0; jb < SEQ; jb += 64) {
    // prefetch next key block's K rows (each lane covers 2 of 64 rows)
    if (jb + 64 < SEQ) {
      __builtin_prefetch(Kh + (size_t)(jb + 64 + lane * 2) * DKH, 0, 1);
      __builtin_prefetch(Vh + (size_t)(lane * 2) * SEQ + jb + 64, 0, 1);
    }

    // ---- batch-load all 8 K B-fragments, then 8 score WMMAs ----
    v16h kb[8];
#pragma unroll
    for (int nt = 0; nt < 4; ++nt) {
      const _Float16* krow = Kh + (size_t)(jb + nt * 16 + l16) * DKH;
      kb[2 * nt]     = ld_b_frag(krow + 0,  hi16);
      kb[2 * nt + 1] = ld_b_frag(krow + 32, hi16);
    }
    v8f s[4];
#pragma unroll
    for (int nt = 0; nt < 4; ++nt) {
      v8f accs = vzero;
      accs = wmma16(qa0, kb[2 * nt],     accs);
      accs = wmma16(qa1, kb[2 * nt + 1], accs);
      s[nt] = accs * 0.125f;                    // 1/sqrt(64)
    }

    // ---- online softmax (rows = vgpr index within 16-lane half) ----
    float mnew[8], corr[8], rsum[8];
#pragma unroll
    for (int r = 0; r < 8; ++r) {
      float v = fmaxf(fmaxf(s[0][r], s[1][r]), fmaxf(s[2][r], s[3][r]));
      v = redmax16(v);
      mnew[r] = fmaxf(m_i[r], v);
      corr[r] = __expf(m_i[r] - mnew[r]);
      rsum[r] = 0.f;
    }
#pragma unroll
    for (int nt = 0; nt < 4; ++nt)
#pragma unroll
      for (int r = 0; r < 8; ++r) {
        float p = __expf(s[nt][r] - mnew[r]);
        rsum[r] += p;
        ldsP[wlocal][(r + hi16 * 8) * 64 + nt * 16 + l16] = (_Float16)p;
      }
#pragma unroll
    for (int r = 0; r < 8; ++r) {
      l_i[r] = l_i[r] * corr[r] + redsum16(rsum[r]);
      m_i[r] = mnew[r];
    }
#pragma unroll
    for (int nt = 0; nt < 4; ++nt)
#pragma unroll
      for (int r = 0; r < 8; ++r) o[nt][r] *= corr[r];

    // ---- issue V global loads BEFORE the DS drain so they overlap ----
    v16h vb[8];
#pragma unroll
    for (int nt = 0; nt < 4; ++nt) {
      const _Float16* vrow = Vh + (size_t)(nt * 16 + l16) * SEQ + jb;
      vb[2 * nt]     = ld_b_frag(vrow + 0,  hi16);
      vb[2 * nt + 1] = ld_b_frag(vrow + 32, hi16);
    }

    // LDS ops from one wave are in-order; wait DS writes, block reordering.
    asm volatile("s_wait_dscnt 0" ::: "memory");

    // ---- reload P in A-fragment layout from LDS ----
    const _Float16* prow = &ldsP[wlocal][l16 * 64];
    v16h pa0 = cat8(*(const v8h*)(prow + hi16 * 8),
                    *(const v8h*)(prow + 16 + hi16 * 8));
    v16h pa1 = cat8(*(const v8h*)(prow + 32 + hi16 * 8),
                    *(const v8h*)(prow + 48 + hi16 * 8));

    // ---- O += P @ V ----
#pragma unroll
    for (int nt = 0; nt < 4; ++nt) {
      o[nt] = wmma16(pa0, vb[2 * nt],     o[nt]);
      o[nt] = wmma16(pa1, vb[2 * nt + 1], o[nt]);
    }
  }

  // ---- normalize and scatter to [B*S, H*Dk] ----
  const int bb = bh >> 4, h = bh & (HEADS - 1);
#pragma unroll
  for (int nt = 0; nt < 4; ++nt)
#pragma unroll
    for (int r = 0; r < 8; ++r) {
      int srow = q0 + r + hi16 * 8;
      size_t idx = ((size_t)(bb * SEQ + srow)) * DMODEL + h * DKH + nt * 16 + l16;
      O[idx] = (_Float16)(o[nt][r] / l_i[r]);
    }
}

// ---------------------------------------------------------------------------
// Kernel 3: output GEMM  out[m,n] = sum_k O[m,k] * Wo[n,k] + b_o[n]  (f32 out)
// ---------------------------------------------------------------------------
__global__ __launch_bounds__(128)
void out_gemm_kernel(const _Float16* __restrict__ X,
                     const _Float16* __restrict__ W,
                     const float* __restrict__ bias,
                     float* __restrict__ Out) {
  const int lane = threadIdx.x & 31;
  const int hi16 = lane >> 4;
  const int l16  = lane & 15;
  const int wid  = (blockIdx.x * blockDim.x + threadIdx.x) >> 5;
  const int m0   = (wid & 127) * 32;
  const int n0   = (wid >> 7)  * 64;

  v8f acc[2][4];
  const v8f vzero = {0.f,0.f,0.f,0.f,0.f,0.f,0.f,0.f};
#pragma unroll
  for (int i = 0; i < 2; ++i)
#pragma unroll
    for (int j = 0; j < 4; ++j) acc[i][j] = vzero;

  gemm32x64_nt(X, W, m0, n0, l16, hi16, acc);

#pragma unroll
  for (int mt = 0; mt < 2; ++mt)
#pragma unroll
    for (int nt = 0; nt < 4; ++nt) {
      int n = n0 + nt * 16 + l16;
      float bn = bias[n];
#pragma unroll
      for (int r = 0; r < 8; ++r) {
        int m = m0 + mt * 16 + r + hi16 * 8;
        Out[(size_t)m * DMODEL + n] = acc[mt][nt][r] + bn;
      }
    }
}

// ---------------------------------------------------------------------------
// Host launch
// ---------------------------------------------------------------------------
extern "C" void kernel_launch(void* const* d_in, const int* in_sizes, int n_in,
                              void* d_out, int out_size, void* d_ws, size_t ws_size,
                              hipStream_t stream) {
  (void)in_sizes; (void)n_in; (void)out_size; (void)ws_size;
  const float* q  = (const float*)d_in[0];
  const float* k  = (const float*)d_in[1];
  const float* v  = (const float*)d_in[2];
  const float* wq = (const float*)d_in[3];
  const float* wk = (const float*)d_in[4];
  const float* wv = (const float*)d_in[5];
  const float* wo = (const float*)d_in[6];
  const float* bo = (const float*)d_in[7];

  char* ws = (char*)d_ws;
  const size_t MB = 1024ull * 1024ull;
  _Float16* Xq = (_Float16*)(ws + 0 * MB);    // 8 MB each activation (f16)
  _Float16* Xk = (_Float16*)(ws + 8 * MB);
  _Float16* Xv = (_Float16*)(ws + 16 * MB);
  _Float16* Wq = (_Float16*)(ws + 24 * MB);   // 2 MB each weight (f16)
  _Float16* Wk = (_Float16*)(ws + 26 * MB);
  _Float16* Wv = (_Float16*)(ws + 28 * MB);
  _Float16* Wo = (_Float16*)(ws + 30 * MB);
  _Float16* Qh = (_Float16*)(ws + 32 * MB);   // [B,H,S,Dk]
  _Float16* Kh = (_Float16*)(ws + 40 * MB);   // [B,H,S,Dk]
  _Float16* Vt = (_Float16*)(ws + 48 * MB);   // [B,H,Dk,S]
  _Float16* Oa = (_Float16*)(ws + 0 * MB);    // reuse X region after projections

  const int nAct4 = (MROWS * DMODEL) / 4;       // 1048576
  const int nW4   = (DMODEL * DMODEL) / 4;      // 262144

  cvt_f32_f16_kernel<<<nAct4 / 256, 256, 0, stream>>>(q,  Xq, nAct4);
  cvt_f32_f16_kernel<<<nAct4 / 256, 256, 0, stream>>>(k,  Xk, nAct4);
  cvt_f32_f16_kernel<<<nAct4 / 256, 256, 0, stream>>>(v,  Xv, nAct4);
  cvt_f32_f16_kernel<<<nW4 / 256,   256, 0, stream>>>(wq, Wq, nW4);
  cvt_f32_f16_kernel<<<nW4 / 256,   256, 0, stream>>>(wk, Wk, nW4);
  cvt_f32_f16_kernel<<<nW4 / 256,   256, 0, stream>>>(wv, Wv, nW4);
  cvt_f32_f16_kernel<<<nW4 / 256,   256, 0, stream>>>(wo, Wo, nW4);

  // 2048 waves = 512 blocks x 4 waves: 128 M-tiles x 16 N-tiles
  proj_gemm_kernel<<<512, 128, 0, stream>>>(Xq, Wq, Qh, 0);
  proj_gemm_kernel<<<512, 128, 0, stream>>>(Xk, Wk, Kh, 0);
  proj_gemm_kernel<<<512, 128, 0, stream>>>(Xv, Wv, Vt, 1);

  // 4096 waves = 1024 blocks x 4 waves: 32 (b,h) x 128 q-tiles
  attention_kernel<<<1024, 128, 0, stream>>>(Qh, Kh, Vt, Oa);

  out_gemm_kernel<<<512, 128, 0, stream>>>(Oa, Wo, bo, (float*)d_out);
}
